// MoEMLP_37933151158753
// MI455X (gfx1250) — matrix operand order repaired and checked by reading, not compiled, not run on hardware
//
#include <hip/hip_runtime.h>
#include <stdint.h>

// ---------------- problem constants ----------------
#define B_   2
#define S_   2048
#define D_   1024
#define E_   8
#define H_   4096
#define NTOK (B_*S_)          // 4096 tokens

// ---------------- WMMA types ----------------
typedef __attribute__((ext_vector_type(16))) __bf16      v16bf;
typedef __attribute__((ext_vector_type(8)))  float       v8f;
typedef __attribute__((ext_vector_type(8)))  unsigned int v8u;

union ABOp { v8u u; v16bf b; uint4 q[2]; };

#if defined(__has_builtin)
#if __has_builtin(__builtin_amdgcn_global_load_async_to_lds_b128)
#define HAVE_ASYNC_COPY 1
#endif
#endif
#ifndef HAVE_ASYNC_COPY
#define HAVE_ASYNC_COPY 0
#endif

#if HAVE_ASYNC_COPY
typedef int v4i __attribute__((__vector_size__(16)));
typedef __attribute__((address_space(1))) v4i* gv4i_p;   // global
typedef __attribute__((address_space(3))) v4i* lv4i_p;   // LDS
#endif

__device__ __forceinline__ uint16_t f2bf(float f) {
  uint32_t u = __float_as_uint(f);
  u += 0x7FFFu + ((u >> 16) & 1u);          // round-to-nearest-even
  return (uint16_t)(u >> 16);
}
__device__ __forceinline__ uint32_t pack2(float lo, float hi) {
  return (uint32_t)f2bf(lo) | ((uint32_t)f2bf(hi) << 16);
}
__device__ __forceinline__ float gelu_exact(float v) {
  return 0.5f * v * (1.0f + erff(v * 0.70710678118654752f));
}

// ---------------- expert-kernel tiling ----------------
#define TM        32            // tokens per block
#define KH        64            // H-chunk per iteration
#define XS_PITCH  1032          // bf16 elems; 2064B row (16B aligned, +4 banks/row)
#define W1S_PITCH 40            // 80B rows (16B aligned)
#define H_PITCH   72            // 144B rows (16B aligned)
#define W2S_PITCH 72
// dynamic-LDS byte offsets
#define XS_OFF     0
#define W1S_OFF    (XS_OFF + TM*XS_PITCH*2)          // 66048
#define W1S_WBYTES (16*W1S_PITCH*2)                  // 1280 per wave
#define HS_OFF     (W1S_OFF + 8*W1S_WBYTES)          // 76288
#define W2S_OFF    (HS_OFF + TM*H_PITCH*2)           // 80896
#define META_OFF   (W2S_OFF + D_*W2S_PITCH*2)        // 228352
#define SMEM_BYTES (META_OFF + TM*12)                // 228736 (< 320KB/WGP)

// ---------------- workspace layout (bytes) ----------------
#define W1T_OFF   0u
#define W1T_BYTES ((size_t)E_*H_*D_*2)               // 67,108,864
#define W2T_OFF   (W1T_OFF + W1T_BYTES)
#define W2T_BYTES ((size_t)E_*D_*H_*2)               // 67,108,864
#define CNT_OFF   (W2T_OFF + W2T_BYTES)
#define LIST_OFF  (CNT_OFF + 256)
#define PLIST_OFF (LIST_OFF  + (size_t)E_*NTOK*4)
#define SLOT_OFF  (PLIST_OFF + (size_t)E_*NTOK*4)
#define YBUF_OFF  (SLOT_OFF  + (size_t)E_*NTOK*4)    // + 8192*1024*4

// ============ fp32 -> bf16 transpose: src[E][R][C] -> dst[E][C][R] ============
__global__ void k_cvt_transpose(const float* __restrict__ src,
                                uint16_t* __restrict__ dst, int R, int C) {
  __shared__ float tile[32][33];
  const int e  = blockIdx.z;
  const int c0 = blockIdx.x * 32, r0 = blockIdx.y * 32;
  const int tx = threadIdx.x, ty = threadIdx.y;      // 32 x 8
  const float* s = src + (size_t)e * R * C;
  uint16_t*    d = dst + (size_t)e * R * C;
#pragma unroll
  for (int i = 0; i < 4; ++i) {
    const int rl = ty + i * 8;
    tile[rl][tx] = s[(size_t)(r0 + rl) * C + c0 + tx];
  }
  __syncthreads();
#pragma unroll
  for (int i = 0; i < 4; ++i) {
    const int cl = ty + i * 8;
    d[(size_t)(c0 + cl) * R + r0 + tx] = f2bf(tile[tx][cl]);
  }
}

__global__ void k_zero_cnt(int* cnt) {
  if (threadIdx.x < E_) cnt[threadIdx.x] = 0;
}

// ============ gating: one wave per token, top-2 + softmax + routing ============
__global__ void k_gate(const float* __restrict__ x, const float* __restrict__ gw,
                       const float* __restrict__ gb, int* __restrict__ cnt,
                       int* __restrict__ lists, float* __restrict__ plist,
                       int* __restrict__ slots) {
  const int t    = blockIdx.x * 8 + (threadIdx.x >> 5);
  const int lane = threadIdx.x & 31;
  float s[E_];
#pragma unroll
  for (int e = 0; e < E_; ++e) s[e] = 0.0f;
  const float* xr = x + (size_t)t * D_;
  for (int d = lane; d < D_; d += 32) {
    const float xv = xr[d];
    const float4* g = (const float4*)(gw + (size_t)d * E_);
    const float4 g0 = g[0], g1 = g[1];
    s[0] += xv * g0.x; s[1] += xv * g0.y; s[2] += xv * g0.z; s[3] += xv * g0.w;
    s[4] += xv * g1.x; s[5] += xv * g1.y; s[6] += xv * g1.z; s[7] += xv * g1.w;
  }
#pragma unroll
  for (int e = 0; e < E_; ++e)
#pragma unroll
    for (int off = 16; off; off >>= 1) s[e] += __shfl_xor(s[e], off, 32);
  if (lane == 0) {
#pragma unroll
    for (int e = 0; e < E_; ++e) s[e] += gb[e];
    int i0 = 0;
#pragma unroll
    for (int e = 1; e < E_; ++e) if (s[e] > s[i0]) i0 = e;
    int i1 = (i0 == 0) ? 1 : 0;
#pragma unroll
    for (int e = 0; e < E_; ++e) if (e != i0 && s[e] > s[i1]) i1 = e;
    const float e1 = expf(s[i1] - s[i0]);
    const float p0 = 1.0f / (1.0f + e1);
    const float p1 = 1.0f - p0;
    int pos0 = atomicAdd(&cnt[i0], 1);
    lists[i0 * NTOK + pos0] = t; plist[i0 * NTOK + pos0] = p0; slots[i0 * NTOK + pos0] = t * 2;
    int pos1 = atomicAdd(&cnt[i1], 1);
    lists[i1 * NTOK + pos1] = t; plist[i1 * NTOK + pos1] = p1; slots[i1 * NTOK + pos1] = t * 2 + 1;
  }
}

// ============ fused expert GEMM->GELU->GEMM (WMMA bf16, f32 accum) ============
__global__ __launch_bounds__(256) void k_expert(
    const float* __restrict__ x,  const uint16_t* __restrict__ w1t,
    const float* __restrict__ b1, const uint16_t* __restrict__ w2t,
    const float* __restrict__ b2, const int* __restrict__ cnt,
    const int* __restrict__ lists, const float* __restrict__ plist,
    const int* __restrict__ slots, float* __restrict__ ybuf) {
  extern __shared__ char smem[];
  const int e    = blockIdx.y;
  const int n    = cnt[e];
  const int row0 = blockIdx.x * TM;
  if (row0 >= n) return;

  const int tid  = threadIdx.x;
  const int lane = tid & 31;
  const int wave = tid >> 5;
  const int hi   = lane >> 4;
  const int l15  = lane & 15;

  int*   tok_s  = (int*)  (smem + META_OFF);
  float* p_s    = (float*)(smem + META_OFF + TM * 4);
  int*   slot_s = (int*)  (smem + META_OFF + TM * 8);
  if (tid < TM) {
    const int idx = row0 + tid;
    const int ci  = idx < n ? idx : (n - 1);
    tok_s[tid]  = lists[e * NTOK + ci];
    p_s[tid]    = (idx < n) ? plist[e * NTOK + ci] : 0.0f;  // padded rows contribute 0
    slot_s[tid] = slots[e * NTOK + ci];
  }
  __syncthreads();

  // ---- gather x rows -> LDS bf16 (row-major, pitch XS_PITCH) ----
  uint32_t* xs32 = (uint32_t*)(smem + XS_OFF);
  for (int j = 0; j < 32; ++j) {
    const int idx = tid + j * 256;               // 8192 float4 pieces
    const int r = idx >> 8, seg = idx & 255;
    const float4 v = *(const float4*)(x + (size_t)tok_s[r] * D_ + seg * 4);
    const int base = (r * XS_PITCH + seg * 4) >> 1;
    xs32[base]     = pack2(v.x, v.y);
    xs32[base + 1] = pack2(v.z, v.w);
  }
  __syncthreads();

  const int mi = wave & 1;                       // stage-1 M subtile
  const int hj = wave >> 1;                      // stage-1 H subtile (0..3)

  const v8f vzero = {0.f, 0.f, 0.f, 0.f, 0.f, 0.f, 0.f, 0.f};
  v8f acc[16];
#pragma unroll
  for (int i = 0; i < 16; ++i) acc[i] = vzero;

  for (int hc = 0; hc < H_ / KH; ++hc) {
    __syncthreads();                             // prior consumers of w2s/h done

    // ---- w2t chunk [D_ rows][KH cols] -> LDS (async when available) ----
    for (int j = 0; j < 32; ++j) {
      const int idx = tid + j * 256;             // 8192 x 16B pieces
      const int d = idx >> 3, seg = idx & 7;
      const uint16_t* gsrc = w2t + (size_t)(e * D_ + d) * H_ + hc * KH + seg * 8;
      char* ldst = smem + W2S_OFF + (d * W2S_PITCH + seg * 8) * 2;
#if HAVE_ASYNC_COPY
      __builtin_amdgcn_global_load_async_to_lds_b128((gv4i_p)gsrc, (lv4i_p)ldst, 0, 0);
#else
      *(uint4*)ldst = *(const uint4*)gsrc;
#endif
    }
    // prefetch next chunk's w1 rows while we compute
    if (hc + 1 < H_ / KH && tid < KH)
      __builtin_prefetch(w1t + (size_t)(e * H_ + (hc + 1) * KH + tid) * D_, 0, 0);

    // ---- stage 1: this wave's h tile (16x16) over K = D_ ----
    v8f acch = vzero;
    const int hrow_base = hc * KH + hj * 16;
    for (int k = 0; k < D_; k += 32) {
      {                                          // wave-private w1 B-tile staging
        const int r = l15, sg = hi;
        const uint16_t* gsrc = w1t + (size_t)(e * H_ + hrow_base + r) * D_ + k + sg * 16;
        const uint4 q0 = *(const uint4*)gsrc;
        const uint4 q1 = *(const uint4*)(gsrc + 8);
        char* wbase = smem + W1S_OFF + wave * W1S_WBYTES;
        *(uint4*)(wbase + (r * W1S_PITCH + sg * 16) * 2)     = q0;
        *(uint4*)(wbase + (r * W1S_PITCH + sg * 16 + 8) * 2) = q1;
      }
      ABOp a, b;
      {  // A: two contiguous 16B runs per lane (words 4*hi.. and 8+4*hi..)
        const char* abase = smem + XS_OFF + (mi * 16 + l15) * XS_PITCH * 2 + k * 2 + hi * 16;
        a.q[0] = *(const uint4*)abase;
        a.q[1] = *(const uint4*)(abase + 32);
      }
      {  // B: two contiguous 16B runs per lane (words 4*hi.. and 8+4*hi..)
        const char* bbase = smem + W1S_OFF + wave * W1S_WBYTES + l15 * W1S_PITCH * 2 + hi * 16;
        b.q[0] = *(const uint4*)bbase;
        b.q[1] = *(const uint4*)(bbase + 32);
      }
      acch = __builtin_amdgcn_wmma_f32_16x16x32_bf16(false, a.b, false, b.b,
                                                     (short)0, acch, false, false);
    }
    // bias + exact GELU -> bf16 h tile in LDS
    const float b1v = b1[e * H_ + hrow_base + l15];
    uint16_t* hs16 = (uint16_t*)(smem + HS_OFF);
#pragma unroll
    for (int v = 0; v < 8; ++v) {
      const float g = gelu_exact(acch[v] + b1v);
      const int row = mi * 16 + v + hi * 8;      // C-layout: M = v + 8*hi
      hs16[row * H_PITCH + hj * 16 + l15] = f2bf(g);
    }
#if HAVE_ASYNC_COPY
#if defined(__has_builtin) && __has_builtin(__builtin_amdgcn_s_wait_asynccnt)
    __builtin_amdgcn_s_wait_asynccnt(0);
#else
    asm volatile("s_wait_asynccnt 0x0" ::: "memory");
#endif
#endif
    __syncthreads();                             // h + w2s visible to all waves

    // ---- stage 2: y[32 x 128-slice] += h[32 x 64] @ w2[64 x 128-slice] ----
#pragma unroll
    for (int kt = 0; kt < 2; ++kt) {
      ABOp a0, a1;
      {
        const char* h0 = smem + HS_OFF + l15 * H_PITCH * 2 + kt * 64 + hi * 16;
        const char* h1 = h0 + 16 * H_PITCH * 2;
        a0.q[0] = *(const uint4*)h0;  a0.q[1] = *(const uint4*)(h0 + 32);
        a1.q[0] = *(const uint4*)h1;  a1.q[1] = *(const uint4*)(h1 + 32);
      }
#pragma unroll
      for (int nt = 0; nt < 8; ++nt) {
        ABOp b;
        const int dcol = wave * 128 + nt * 16 + l15;
        const char* bbase = smem + W2S_OFF + dcol * W2S_PITCH * 2 + kt * 64 + hi * 16;
        b.q[0] = *(const uint4*)bbase;
        b.q[1] = *(const uint4*)(bbase + 32);
        acc[nt]     = __builtin_amdgcn_wmma_f32_16x16x32_bf16(false, a0.b, false, b.b,
                                                              (short)0, acc[nt], false, false);
        acc[8 + nt] = __builtin_amdgcn_wmma_f32_16x16x32_bf16(false, a1.b, false, b.b,
                                                              (short)0, acc[8 + nt], false, false);
      }
    }
  }

  // ---- epilogue: y = p * (acc + b2) scattered to per-assignment rows ----
#pragma unroll
  for (int mi2 = 0; mi2 < 2; ++mi2) {
#pragma unroll
    for (int nt = 0; nt < 8; ++nt) {
      const int dcol = wave * 128 + nt * 16 + l15;
      const float b2v = b2[e * D_ + dcol];
#pragma unroll
      for (int v = 0; v < 8; ++v) {
        const int row = mi2 * 16 + v + hi * 8;
        const float val = p_s[row] * (acc[mi2 * 8 + nt][v] + b2v);
        ybuf[(size_t)slot_s[row] * D_ + dcol] = val;
      }
    }
  }
}

// ============ deterministic combine: out[t] = y[2t] + y[2t+1] ============
__global__ void k_combine(const float* __restrict__ ybuf, float* __restrict__ out) {
  const int t = blockIdx.x;
  const int d = threadIdx.x * 4;
  const float4 a = *(const float4*)(ybuf + (size_t)(2 * t) * D_ + d);
  const float4 b = *(const float4*)(ybuf + (size_t)(2 * t + 1) * D_ + d);
  float4 r;
  r.x = a.x + b.x; r.y = a.y + b.y; r.z = a.z + b.z; r.w = a.w + b.w;
  *(float4*)(out + (size_t)t * D_ + d) = r;
}

// ---------------- host launcher ----------------
extern "C" void kernel_launch(void* const* d_in, const int* in_sizes, int n_in,
                              void* d_out, int out_size, void* d_ws, size_t ws_size,
                              hipStream_t stream) {
  (void)in_sizes; (void)n_in; (void)out_size; (void)ws_size;
  const float* x  = (const float*)d_in[0];
  const float* w1 = (const float*)d_in[1];
  const float* b1 = (const float*)d_in[2];
  const float* w2 = (const float*)d_in[3];
  const float* b2 = (const float*)d_in[4];
  const float* gw = (const float*)d_in[5];
  const float* gb = (const float*)d_in[6];
  float* out = (float*)d_out;

  char* ws = (char*)d_ws;
  uint16_t* w1t   = (uint16_t*)(ws + W1T_OFF);
  uint16_t* w2t   = (uint16_t*)(ws + W2T_OFF);
  int*      cntp  = (int*)     (ws + CNT_OFF);
  int*      lists = (int*)     (ws + LIST_OFF);
  float*    plist = (float*)   (ws + PLIST_OFF);
  int*      slotsp= (int*)     (ws + SLOT_OFF);
  float*    ybuf  = (float*)   (ws + YBUF_OFF);

  (void)hipFuncSetAttribute(reinterpret_cast<const void*>(k_expert),
                            hipFuncAttributeMaxDynamicSharedMemorySize, SMEM_BYTES);

  const dim3 tb(32, 8);
  // w1 [E][D][H] -> w1t [E][H][D] ; w2 [E][H][D] -> w2t [E][D][H]
  k_cvt_transpose<<<dim3(H_ / 32, D_ / 32, E_), tb, 0, stream>>>(w1, w1t, D_, H_);
  k_cvt_transpose<<<dim3(D_ / 32, H_ / 32, E_), tb, 0, stream>>>(w2, w2t, H_, D_);
  k_zero_cnt<<<1, 32, 0, stream>>>(cntp);
  k_gate<<<NTOK / 8, 256, 0, stream>>>(x, gw, gb, cntp, lists, plist, slotsp);
  k_expert<<<dim3(NTOK / TM, E_), 256, SMEM_BYTES, stream>>>(
      x, w1t, b1, w2t, b2, cntp, lists, plist, slotsp, ybuf);
  k_combine<<<NTOK, 256, 0, stream>>>(ybuf, out);
}
